// ObjectRepresentationSNN_6768868459237
// MI455X (gfx1250) — matrix operand order repaired and checked by reading, not compiled, not run on hardware
//
#include <hip/hip_runtime.h>
#include <stdint.h>

// ---- problem constants (match reference) ----
#define B_   32
#define N_   256
#define P_   3072
#define T_   200
#define D_   11
#define BN   (B_*N_)
#define BP   (B_*P_)

typedef __attribute__((ext_vector_type(16))) _Float16 v16h;
typedef __attribute__((ext_vector_type(8)))  _Float16 v8h;
typedef __attribute__((ext_vector_type(8)))  float    v8f;

// Generic pointers into LDS carry the wave-relative LDS offset in their low 32
// bits (ISA 10.2, LDS aperture).
__device__ __forceinline__ uint32_t lds_offset_of(const void* p) {
  return (uint32_t)(uintptr_t)p;
}

// ---------------- WMMA fragment loaders ----------------
// A fragment (16x32 f16, row-major source): lanes 0-15 hold M=lane with
// K = k0+{0..7,16..23}; lanes 16-31 hold K = k0+{8..15,24..31}.
__device__ __forceinline__ v16h load_a_f16(const _Float16* __restrict__ Ah, int lda,
                                           int mbase, int k0, int lane) {
  int m   = mbase + (lane & 15);
  int off = (lane < 16) ? 0 : 8;
  const _Float16* row = Ah + (size_t)m * lda + k0 + off;
  v8h lo = *(const v8h*)(row);        // 16B aligned -> global_load_b128
  v8h hi = *(const v8h*)(row + 16);
  v16h f;
#pragma unroll
  for (int i = 0; i < 8; i++) { f[i] = lo[i]; f[8 + i] = hi[i]; }
  return f;
}

__device__ __forceinline__ v8f wmma_acc(v16h a, v16h b, v8f c) {
  return __builtin_amdgcn_wmma_f32_16x16x32_f16(false, a, false, b, (short)0, c,
                                                false, false);
}

// Direct GEMM, both M-tiles per wave: one B panel feeds two WMMAs.
// Packed B panel: per (ntile, kstep) block, 32 lanes x 16 f16 contiguous.
__device__ __forceinline__ void gemm2m_f16(const _Float16* __restrict__ Ah, int lda,
                                           const _Float16* __restrict__ Bp, int ksteps,
                                           int nt, int lane, v8f& c0, v8f& c1) {
  const _Float16* bsrc = Bp + ((size_t)nt * ksteps * 32 + lane) * 16;
  for (int kb = 0; kb < ksteps; kb += 2) {
    __builtin_prefetch(bsrc + 2 * 512, 0, 1);   // -> global_prefetch_b8
    v16h b0 = *(const v16h*)(bsrc);
    v16h b1 = *(const v16h*)(bsrc + 512);
    v16h a00 = load_a_f16(Ah, lda, 0,  (kb + 0) * 32, lane);
    c0 = wmma_acc(a00, b0, c0);
    v16h a10 = load_a_f16(Ah, lda, 16, (kb + 0) * 32, lane);
    c1 = wmma_acc(a10, b0, c1);
    v16h a01 = load_a_f16(Ah, lda, 0,  (kb + 1) * 32, lane);
    c0 = wmma_acc(a01, b1, c0);
    v16h a11 = load_a_f16(Ah, lda, 16, (kb + 1) * 32, lane);
    c1 = wmma_acc(a11, b1, c1);
    bsrc += 1024;
  }
}

// Staged GEMM, both M-tiles per wave: B panels double-buffered in per-wave LDS
// via gfx1250 async global->LDS copies (ASYNCcnt), consumed with ds_load_b128.
__device__ __forceinline__ void gemm2m_staged(const _Float16* __restrict__ Ah, int lda,
                                              const _Float16* __restrict__ Bp, int ksteps,
                                              _Float16* shbase,  // 1024 f16 per wave
                                              int nt, int lane, v8f& c0, v8f& c1) {
  uint64_t gsrc = (uint64_t)(uintptr_t)(Bp + ((size_t)nt * ksteps * 32 + lane) * 16);
  uint32_t lds0 = lds_offset_of(shbase + lane * 16);
  uint32_t lds1 = lds_offset_of(shbase + 512 + lane * 16);
  // prologue: stage kb=0 into buffer 0 (2 x 16B per lane = 1KB panel per wave)
  asm volatile("global_load_async_to_lds_b128 %0, %1, off"
               :: "v"(lds0), "v"(gsrc) : "memory");
  asm volatile("global_load_async_to_lds_b128 %0, %1, off offset:16"
               :: "v"(lds0), "v"(gsrc) : "memory");
  for (int kb = 0; kb < ksteps; kb++) {
    uint64_t gnext   = gsrc + 1024;               // next 512-f16 panel
    uint32_t ldsNext = (kb & 1) ? lds0 : lds1;
    if (kb + 1 < ksteps) {
      asm volatile("global_load_async_to_lds_b128 %0, %1, off"
                   :: "v"(ldsNext), "v"(gnext) : "memory");
      asm volatile("global_load_async_to_lds_b128 %0, %1, off offset:16"
                   :: "v"(ldsNext), "v"(gnext) : "memory");
      asm volatile("s_wait_asynccnt 0x2" ::: "memory");  // current panel landed
    } else {
      asm volatile("s_wait_asynccnt 0x0" ::: "memory");
    }
    const _Float16* cur = (kb & 1) ? (shbase + 512) : shbase;
    v16h b = *(const v16h*)(cur + lane * 16);     // ds_load_b128 x2
    v16h a0 = load_a_f16(Ah, lda, 0,  kb * 32, lane);
    c0 = wmma_acc(a0, b, c0);
    v16h a1 = load_a_f16(Ah, lda, 16, kb * 32, lane);
    c1 = wmma_acc(a1, b, c1);
    gsrc = gnext;
  }
}

// ---------------- kernels ----------------

// Pack f32 row-major B[K x Nout] into f16 WMMA panels. One thread per
// (block, lane); blocks enumerated as nt*ksteps + kb.
__global__ void pack_b_kernel(const float* __restrict__ Bw, int ldb, int ksteps,
                              _Float16* __restrict__ Bp, int total) {
  int t = blockIdx.x * blockDim.x + threadIdx.x;
  if (t >= total) return;
  int lane = t & 31;
  int blk  = t >> 5;
  int kb   = blk % ksteps;
  int nt   = blk / ksteps;
  int n    = nt * 16 + (lane & 15);
  int k0   = kb * 32 + ((lane >> 4) << 3);
  _Float16* dst = Bp + ((size_t)blk * 32 + lane) * 16;
#pragma unroll
  for (int i = 0; i < 16; i++)
    dst[i] = (_Float16)Bw[(size_t)(k0 + i + ((i >> 3) << 3)) * ldb + n];
}

// Out(f32) = tanh(Ah @ Bpacked); 8 waves/block, one N-tile (both M-tiles)/wave.
__global__ void gemm_tanh_kernel(const _Float16* __restrict__ Ah, int lda,
                                 const _Float16* __restrict__ Bp, int ksteps,
                                 float* __restrict__ Out, int ldo) {
  int lane = threadIdx.x & 31;
  int nt   = blockIdx.x * (blockDim.x >> 5) + (threadIdx.x >> 5);
  v8f c0 = {};
  v8f c1 = {};
  gemm2m_f16(Ah, lda, Bp, ksteps, nt, lane, c0, c1);
  int n  = nt * 16 + (lane & 15);
  int m0 = (lane >> 4) << 3;            // C/D: VGPR r <-> M = r + 8*(lane>=16)
#pragma unroll
  for (int r = 0; r < 8; r++) {
    Out[(size_t)(m0 + r) * ldo + n]      = tanhf(c0[r]);
    Out[(size_t)(16 + m0 + r) * ldo + n] = tanhf(c1[r]);
  }
}

// gamma(f32) = tanh(theta @ W_ro); gateh(f16) = gamma*0.5*(1+cos(theta-delayed))
__global__ void readout_gate_kernel(const _Float16* __restrict__ thetah,
                                    const float* __restrict__ theta,
                                    const _Float16* __restrict__ Wrop,
                                    const float* __restrict__ delayed,
                                    float* __restrict__ gamma,
                                    _Float16* __restrict__ gateh) {
  int lane = threadIdx.x & 31;
  int nt   = blockIdx.x * (blockDim.x >> 5) + (threadIdx.x >> 5);
  v8f c0 = {};
  v8f c1 = {};
  gemm2m_f16(thetah, N_, Wrop, N_ / 32, nt, lane, c0, c1);
  int n  = nt * 16 + (lane & 15);
  int m0 = (lane >> 4) << 3;
#pragma unroll
  for (int r = 0; r < 8; r++) {
    int i0 = (m0 + r) * N_ + n;
    int i1 = (16 + m0 + r) * N_ + n;
    float g0 = tanhf(c0[r]);
    float g1 = tanhf(c1[r]);
    gamma[i0] = g0;
    gamma[i1] = g1;
    gateh[i0] = (_Float16)(g0 * 0.5f * (1.0f + cosf(theta[i0] - delayed[i0])));
    gateh[i1] = (_Float16)(g1 * 0.5f * (1.0f + cosf(theta[i1] - delayed[i1])));
  }
}

// cur = gate@W_gate + 0.1*spk@W_rec ; mem = 0.9*mem*(1-spk) + cur ; spk' = H(mem-1)
__global__ void __launch_bounds__(256)
spike_update_kernel(const _Float16* __restrict__ gateh,
                    const _Float16* __restrict__ Wgp,
                    const _Float16* __restrict__ spkh_in,
                    const _Float16* __restrict__ Wrp,
                    float* __restrict__ mem,
                    _Float16* __restrict__ spkh_out) {
  __shared__ _Float16 sh[8 * 1024];                 // 8 waves x 2KB double buffer
  int lane = threadIdx.x & 31;
  int wv   = threadIdx.x >> 5;
  int nt   = blockIdx.x * (blockDim.x >> 5) + wv;
  v8f g0 = {}, g1 = {};                             // gate @ W_gate
  v8f r0 = {}, r1 = {};                             // spk  @ W_rec (staged)
  gemm2m_f16(gateh, N_, Wgp, N_ / 32, nt, lane, g0, g1);
  gemm2m_staged(spkh_in, P_, Wrp, P_ / 32, sh + wv * 1024, nt, lane, r0, r1);
  int n  = nt * 16 + (lane & 15);
  int m0 = (lane >> 4) << 3;
#pragma unroll
  for (int r = 0; r < 8; r++) {
    size_t i0 = (size_t)(m0 + r) * P_ + n;
    size_t i1 = (size_t)(16 + m0 + r) * P_ + n;
    float cur0 = g0[r] + 0.1f * r0[r];
    float cur1 = g1[r] + 0.1f * r1[r];
    float so0  = (float)spkh_in[i0];                // spikes are exactly 0/1
    float so1  = (float)spkh_in[i1];
    float mn0  = 0.9f * mem[i0] * (1.0f - so0) + cur0;
    float mn1  = 0.9f * mem[i1] * (1.0f - so1) + cur1;
    mem[i0] = mn0;
    mem[i1] = mn1;
    spkh_out[i0] = (_Float16)((mn0 - 1.0f > 0.0f) ? 1.0f : 0.0f);
    spkh_out[i1] = (_Float16)((mn1 - 1.0f > 0.0f) ? 1.0f : 0.0f);
  }
}

__device__ __forceinline__ float block_reduce256(float v, float* red) {
  int t = threadIdx.x;
  red[t] = v;
  __syncthreads();
  for (int off = 128; off > 0; off >>= 1) {
    if (t < off) red[t] += red[t + off];
    __syncthreads();
  }
  float r = red[0];
  __syncthreads();
  return r;
}

// One block per batch row; rank-1 coupling => 4 block reductions over N=256.
__global__ void kuramoto_kernel(float* __restrict__ theta,
                                _Float16* __restrict__ thetah,
                                const float* __restrict__ gamma,
                                const float* __restrict__ a,
                                const float* __restrict__ omega,
                                float* __restrict__ ring,
                                float* __restrict__ delayed,
                                int s) {
  __shared__ float red[N_];
  int b = blockIdx.x, n = threadIdx.x;
  int idx = b * N_ + n;
  float th = theta[idx];
  float av = a[idx];
  float sn = sinf(th), cs = cosf(th);
  float S  = block_reduce256(sn, red);
  float C  = block_reduce256(cs, red);
  float AS = block_reduce256(av * sn, red);
  float AC = block_reduce256(av * cs, red);
  float base     = cs * S - sn * C;
  float weighted = 0.5f * (cs * av * AS - sn * av * AC);   // ALPHA_FIX = 0.5
  float dth = omega[n] + (1.0f / N_) * (base + weighted) + 0.5f * gamma[idx];
  float thn = th + 0.1f * dth;
  theta[idx]  = thn;
  thetah[idx] = (_Float16)thn;
  int slot = s % D_;                      // slot holds theta_{s-11} (theta0 pre-wrap)
  delayed[idx] = ring[slot * BN + idx];   // read old value = buf[0]
  ring[slot * BN + idx] = thn;            // append theta_s
}

__global__ void init_state_kernel(const float* __restrict__ theta0,
                                  const float* __restrict__ x,
                                  float* theta, _Float16* thetah,
                                  float* ring, float* delayed,
                                  _Float16* gateh, float* a,
                                  float* mem, _Float16* spkAh, _Float16* spkBh,
                                  float* rate, _Float16* xh) {
  int i = blockIdx.x * blockDim.x + threadIdx.x;
  if (i < BN) {
    float t0 = theta0[i];
    theta[i]   = t0;
    thetah[i]  = (_Float16)t0;
    delayed[i] = t0;
    gateh[i]   = (_Float16)0.0f;
    a[i]       = 0.0f;
#pragma unroll
    for (int d = 0; d < D_; d++) ring[d * BN + i] = t0;
  }
  if (i < BP) {
    mem[i]   = 0.0f;
    spkAh[i] = (_Float16)0.0f;
    spkBh[i] = (_Float16)0.0f;
    rate[i]  = 0.0f;
    xh[i]    = (_Float16)x[i];
  }
}

__global__ void rate_accum_kernel(float* __restrict__ rate,
                                  const _Float16* __restrict__ spkh) {
  int i = blockIdx.x * blockDim.x + threadIdx.x;
  if (i < BP) rate[i] += (float)spkh[i];
}

// logits[b,c] = (1/150) * sum_p rate[b,p] * Wcls[p,c] + bcls[c]
__global__ void logits_kernel(const float* __restrict__ rate,
                              const float* __restrict__ Wcls,
                              const float* __restrict__ bcls,
                              float* __restrict__ out) {
  __shared__ float red[256];
  int b = blockIdx.x / 10;
  int c = blockIdx.x % 10;
  int t = threadIdx.x;
  float s = 0.0f;
  for (int p = t; p < P_; p += 256)
    s += rate[(size_t)b * P_ + p] * Wcls[(size_t)p * 10 + c];
  red[t] = s;
  __syncthreads();
  for (int off = 128; off > 0; off >>= 1) {
    if (t < off) red[t] += red[t + off];
    __syncthreads();
  }
  if (t == 0) out[b * 10 + c] = red[0] * (1.0f / 150.0f) + bcls[c];
}

// ---------------- host orchestration ----------------
extern "C" void kernel_launch(void* const* d_in, const int* in_sizes, int n_in,
                              void* d_out, int out_size, void* d_ws, size_t ws_size,
                              hipStream_t stream) {
  (void)in_sizes; (void)n_in; (void)out_size; (void)ws_size;
  const float* x      = (const float*)d_in[0];
  const float* theta0 = (const float*)d_in[1];
  const float* Winit  = (const float*)d_in[2];
  const float* omega  = (const float*)d_in[3];
  const float* Wro    = (const float*)d_in[4];
  const float* Wgate  = (const float*)d_in[5];
  const float* Wrec   = (const float*)d_in[6];
  const float* Wfb    = (const float*)d_in[7];
  const float* Wcls   = (const float*)d_in[8];
  const float* bcls   = (const float*)d_in[9];
  float* out = (float*)d_out;

  // ---- workspace carve-up: f32 state first, then f16 area (32B aligned) ----
  float* wf      = (float*)d_ws;
  float* theta   = wf; wf += BN;
  float* gamma   = wf; wf += BN;
  float* a       = wf; wf += BN;
  float* delayed = wf; wf += BN;
  float* mem     = wf; wf += BP;
  float* rate    = wf; wf += BP;
  float* ring    = wf; wf += D_ * BN;

  _Float16* wh    = (_Float16*)wf;
  _Float16* thetah = wh; wh += BN;
  _Float16* gateh  = wh; wh += BN;
  _Float16* xh     = wh; wh += BP;
  _Float16* spkAh  = wh; wh += BP;
  _Float16* spkBh  = wh; wh += BP;
  // packed weights: (Nout/16)*(K/32)*512 f16 each
  _Float16* Wip  = wh; wh += (N_/16) * (P_/32) * 512;   // W_init  [P,N]
  _Float16* Wrop = wh; wh += (N_/16) * (N_/32) * 512;   // W_ro    [N,N]
  _Float16* Wgp  = wh; wh += (P_/16) * (N_/32) * 512;   // W_gate  [N,P]
  _Float16* Wrp  = wh; wh += (P_/16) * (P_/32) * 512;   // W_rec   [P,P]
  _Float16* Wfp  = wh; wh += (N_/16) * (P_/32) * 512;   // W_fb    [P,N]

  init_state_kernel<<<(BP + 255) / 256, 256, 0, stream>>>(
      theta0, x, theta, thetah, ring, delayed, gateh, a, mem, spkAh, spkBh,
      rate, xh);

  // pack weights into WMMA-native f16 panels
  {
    int tot;
    tot = (N_/16) * (P_/32) * 32;
    pack_b_kernel<<<(tot + 255) / 256, 256, 0, stream>>>(Winit, N_, P_/32, Wip, tot);
    tot = (N_/16) * (N_/32) * 32;
    pack_b_kernel<<<(tot + 255) / 256, 256, 0, stream>>>(Wro, N_, N_/32, Wrop, tot);
    tot = (P_/16) * (N_/32) * 32;
    pack_b_kernel<<<(tot + 255) / 256, 256, 0, stream>>>(Wgate, P_, N_/32, Wgp, tot);
    tot = (P_/16) * (P_/32) * 32;
    pack_b_kernel<<<(tot + 255) / 256, 256, 0, stream>>>(Wrec, P_, P_/32, Wrp, tot);
    tot = (N_/16) * (P_/32) * 32;
    pack_b_kernel<<<(tot + 255) / 256, 256, 0, stream>>>(Wfb, N_, P_/32, Wfp, tot);
  }

  // gamma0 = tanh(x @ W_init): 16 N-tiles -> 2 blocks x 8 waves
  gemm_tanh_kernel<<<2, 256, 0, stream>>>(xh, P_, Wip, P_/32, gamma, N_);

  _Float16* spk_cur = spkAh;
  _Float16* spk_nxt = spkBh;
  for (int s = 1; s <= T_; s++) {
    kuramoto_kernel<<<B_, N_, 0, stream>>>(theta, thetah, gamma, a, omega,
                                           ring, delayed, s);
    if (s % 5 == 0)  // readout steps: s = 5,10,...,200
      readout_gate_kernel<<<2, 256, 0, stream>>>(thetah, theta, Wrop, delayed,
                                                 gamma, gateh);
    if (s > 1 && ((s - 1) % 5) == 0) {  // spike steps: s = 6,11,...,196
      spike_update_kernel<<<24, 256, 0, stream>>>(gateh, Wgp, spk_cur, Wrp,
                                                  mem, spk_nxt);
      _Float16* tmp = spk_cur; spk_cur = spk_nxt; spk_nxt = tmp;
      // a = tanh(spk @ W_fb)
      gemm_tanh_kernel<<<2, 256, 0, stream>>>(spk_cur, P_, Wfp, P_/32, a, N_);
    }
    if (s >= 51)
      rate_accum_kernel<<<(BP + 255) / 256, 256, 0, stream>>>(rate, spk_cur);
  }
  logits_kernel<<<B_ * 10, 256, 0, stream>>>(rate, Wcls, bcls, out);
}
